// DeformConv1d_53214644798136
// MI455X (gfx1250) — matrix operand order repaired and checked by reading, not compiled, not run on hardware
//
#include <hip/hip_runtime.h>
#include <hip/hip_bf16.h>
#include <math.h>

typedef _Float16 v16h __attribute__((ext_vector_type(16)));
typedef _Float16 v8h  __attribute__((ext_vector_type(8)));
typedef float    v8f  __attribute__((ext_vector_type(8)));

#define BB    16
#define CC    64
#define COUT  64
#define LL    16384
#define KK    3
#define KC    192          // C*K contraction length
#define NT    64           // L-tile per workgroup
#define VSTR  200          // val_lds row stride in halves (400B, 16B-aligned)

__global__ __launch_bounds__(256)
void deform_conv1d_fused(const float* __restrict__ x,
                         const float* __restrict__ w_off_dw,
                         const float* __restrict__ b_off_dw,
                         const float* __restrict__ w_off_pw,
                         const float* __restrict__ b_off_pw,
                         const float* __restrict__ w_msk_dw,
                         const float* __restrict__ b_msk_dw,
                         const float* __restrict__ w_msk_pw,
                         const float* __restrict__ b_msk_pw,
                         const float* __restrict__ weight,
                         const float* __restrict__ bias,
                         float* __restrict__ out)
{
    __shared__ _Float16 wlds[COUT * KC];       // W as f16, [m][kc]
    __shared__ _Float16 val_lds[NT * VSTR];    // val as f16, [n][kc]
    __shared__ float pOff[4][NT][3];
    __shared__ float pMsk[4][NT][3];
    __shared__ float offL[NT][3];
    __shared__ float mskL[NT][3];

    const int tid = threadIdx.x;
    const int b   = blockIdx.y;
    const int l0  = blockIdx.x * NT;

    // ---- weights -> f16 LDS (global layout [o][c][k] == [m][kc]) ----
    #pragma unroll 4
    for (int i = tid; i < COUT * KC; i += 256)
        wlds[i] = (_Float16)weight[i];

    // ---- Phase A1: offset/mask partial sums (4 channel-groups x 64 pos) ----
    {
        const int n  = tid & (NT - 1);
        const int cg = tid >> 6;           // 0..3 -> channels cg*16..cg*16+15
        const int l  = l0 + n;
        float aO0 = 0.f, aO1 = 0.f, aO2 = 0.f;
        float aM0 = 0.f, aM1 = 0.f, aM2 = 0.f;
        for (int c = cg * 16; c < cg * 16 + 16; ++c) {
            const float* xc = x + ((size_t)b * CC + c) * LL;
            float dO = b_off_dw[c];
            float dM = b_msk_dw[c];
            #pragma unroll
            for (int t = 0; t < 7; ++t) {
                const int pos = l - 3 + t;
                const float xv = (pos >= 0 && pos < LL) ? xc[pos] : 0.f;
                dO = fmaf(w_off_dw[c * 7 + t], xv, dO);
                dM = fmaf(w_msk_dw[c * 7 + t], xv, dM);
            }
            aO0 = fmaf(w_off_pw[0 * CC + c], dO, aO0);
            aO1 = fmaf(w_off_pw[1 * CC + c], dO, aO1);
            aO2 = fmaf(w_off_pw[2 * CC + c], dO, aO2);
            aM0 = fmaf(w_msk_pw[0 * CC + c], dM, aM0);
            aM1 = fmaf(w_msk_pw[1 * CC + c], dM, aM1);
            aM2 = fmaf(w_msk_pw[2 * CC + c], dM, aM2);
        }
        pOff[cg][n][0] = aO0; pOff[cg][n][1] = aO1; pOff[cg][n][2] = aO2;
        pMsk[cg][n][0] = aM0; pMsk[cg][n][1] = aM1; pMsk[cg][n][2] = aM2;
    }
    __syncthreads();

    // ---- Phase A2: reduce partials, bias, sigmoid ----
    if (tid < NT) {
        #pragma unroll
        for (int j = 0; j < 3; ++j) {
            float o = b_off_pw[j] + pOff[0][tid][j] + pOff[1][tid][j]
                                  + pOff[2][tid][j] + pOff[3][tid][j];
            float m = b_msk_pw[j] + pMsk[0][tid][j] + pMsk[1][tid][j]
                                  + pMsk[2][tid][j] + pMsk[3][tid][j];
            offL[tid][j] = o;
            mskL[tid][j] = 1.f / (1.f + expf(-m));
        }
    }
    __syncthreads();

    // ---- Phase A3: deformable gather -> val_lds[n][c*3+k] (f16) ----
    for (int idx = tid; idx < CC * NT; idx += 256) {
        const int c = idx >> 6;
        const int n = idx & (NT - 1);
        const float* xc = x + ((size_t)b * CC + c) * LL;
        const int l = l0 + n;
        #pragma unroll
        for (int k = 0; k < KK; ++k) {
            const float p  = (float)(l - 1 + k) + offL[n][k];
            const float fl = floorf(p);
            const int   i0 = (int)fl;
            const int   i1 = i0 + 1;
            const float f  = p - fl;
            const float x0 = (i0 >= 0 && i0 < LL) ? xc[i0] : 0.f;
            const float x1 = (i1 >= 0 && i1 < LL) ? xc[i1] : 0.f;
            const float v  = ((1.f - f) * x0 + f * x1) * mskL[n][k];
            val_lds[n * VSTR + c * 3 + k] = (_Float16)v;
        }
    }
    __syncthreads();

    // ---- Phase B: out[64 x 64tile] = W[64x192] x val[192x64] via WMMA ----
    const int wave = tid >> 5;       // 8 waves
    const int lane = tid & 31;
    const int row  = lane & 15;      // M (for A) / N (for B,D)
    const int half = lane >> 4;      // 0/1
    const int mt   = wave & 3;       // M tile 0..3
    const int ng   = wave >> 2;      // N tile group: tiles ng*2, ng*2+1

    // A fragments: ISA 16-bit A 16x32 layout
    //   half j<8 : K = kt*32 + 8*half + j
    //   half j>=8: K = kt*32 + 8*half + 16 + (j-8)
    v16h afrag[6];
    #pragma unroll
    for (int kt = 0; kt < 6; ++kt) {
        const _Float16* abase = &wlds[(mt * 16 + row) * KC + kt * 32 + half * 8];
        const v8h lo = *(const v8h*)(abase);
        const v8h hi = *(const v8h*)(abase + 16);
        v16h a;
        #pragma unroll
        for (int e = 0; e < 8; ++e) { a[e] = lo[e]; a[e + 8] = hi[e]; }
        afrag[kt] = a;
    }

    #pragma unroll
    for (int nti = 0; nti < 2; ++nti) {
        const int nt = ng * 2 + nti;
        v8f acc = {};
        #pragma unroll
        for (int kt = 0; kt < 6; ++kt) {
            // B fragment: j<8 K=kt*32+8*half+j, j>=8 K=kt*32+8*half+16+(j-8)
            const _Float16* bbase =
                &val_lds[(nt * 16 + row) * VSTR + kt * 32 + half * 8];
            const v8h blo = *(const v8h*)(bbase);
            const v8h bhi = *(const v8h*)(bbase + 16);
            v16h bf;
            #pragma unroll
            for (int e = 0; e < 8; ++e) { bf[e] = blo[e]; bf[e + 8] = bhi[e]; }
            acc = __builtin_amdgcn_wmma_f32_16x16x32_f16(
                false, afrag[kt], false, bf, (short)0, acc, false, false);
        }
        // D layout: VGPR r -> M = mt*16 + r + 8*half, N = nt*16 + row
        const int ncol = l0 + nt * 16 + row;
        #pragma unroll
        for (int r = 0; r < 8; ++r) {
            const int m = mt * 16 + r + 8 * half;
            out[((size_t)b * COUT + m) * LL + ncol] = acc[r] + bias[m];
        }
    }
}

extern "C" void kernel_launch(void* const* d_in, const int* in_sizes, int n_in,
                              void* d_out, int out_size, void* d_ws, size_t ws_size,
                              hipStream_t stream) {
    (void)in_sizes; (void)n_in; (void)out_size; (void)d_ws; (void)ws_size;
    const float* x        = (const float*)d_in[0];
    const float* w_off_dw = (const float*)d_in[1];
    const float* b_off_dw = (const float*)d_in[2];
    const float* w_off_pw = (const float*)d_in[3];
    const float* b_off_pw = (const float*)d_in[4];
    const float* w_msk_dw = (const float*)d_in[5];
    const float* b_msk_dw = (const float*)d_in[6];
    const float* w_msk_pw = (const float*)d_in[7];
    const float* b_msk_pw = (const float*)d_in[8];
    const float* weight   = (const float*)d_in[9];
    const float* bias     = (const float*)d_in[10];
    float* out = (float*)d_out;

    dim3 grid(LL / NT, BB, 1);   // (256, 16)
    dim3 block(256, 1, 1);
    deform_conv1d_fused<<<grid, block, 0, stream>>>(
        x, w_off_dw, b_off_dw, w_off_pw, b_off_pw,
        w_msk_dw, b_msk_dw, w_msk_pw, b_msk_pw, weight, bias, out);
}
